// FP8Module_53309134078229
// MI455X (gfx1250) — 8-bit matrix operand order repaired and checked by reading, compiled, not run on hardware
//
#include <hip/hip_runtime.h>
#include <hip/hip_bf16.h>

#define GEMM_M 8192
#define GEMM_N 4096
#define GEMM_K 4096
#define KSTEP  128
#define KSTEPS (GEMM_K / KSTEP)

typedef __attribute__((ext_vector_type(16))) int           v16i;
typedef __attribute__((ext_vector_type(8)))  float         v8f;
typedef __attribute__((ext_vector_type(4))) unsigned int   u32x4;
typedef __attribute__((ext_vector_type(2))) unsigned int   u32x2;
typedef __attribute__((ext_vector_type(8)))  int           i32x8;
typedef __attribute__((ext_vector_type(4)))  int           i32x4;

__device__ __forceinline__ unsigned int f32_to_e4m3(float x) {
  unsigned int u = __float_as_uint(x);
  unsigned int s = (u >> 24) & 0x80u;
  unsigned int a = u & 0x7FFFFFFFu;
  if (a >= 0x7F800000u) return s | 0x7Fu;
  int ue = (int)(a >> 23) - 127;
  unsigned int m = a & 0x7FFFFFu;
  if (ue >= -6) {
    unsigned int mant = m >> 20;
    unsigned int rem  = m & 0xFFFFFu;
    mant += (rem > 0x80000u) || (rem == 0x80000u && (mant & 1u));
    if (mant == 8u) { mant = 0u; ue += 1; }
    if (ue > 8) return s | 0x7Eu;
    return s | ((unsigned)(ue + 7) << 3) | mant;
  }
  if (ue < -10) return s;
  unsigned int full = m | 0x800000u;
  int sh = 14 - ue;
  unsigned int d    = full >> sh;
  unsigned int r    = full & ((1u << sh) - 1u);
  unsigned int half = 1u << (sh - 1);
  d += (r > half) || (r == half && (d & 1u));
  if (d >= 8u) return s | 0x08u;
  return s | d;
}

__global__ void quantize_e4m3_kernel(const float* __restrict__ in,
                                     unsigned char* __restrict__ out, int n16) {
  int t = blockIdx.x * blockDim.x + threadIdx.x;
  if (t >= n16) return;
  const float4* p = (const float4*)in + (size_t)t * 4;
  unsigned int w[4];
#pragma unroll
  for (int j = 0; j < 4; ++j) {
    float4 f = p[j];
    w[j] = f32_to_e4m3(f.x) | (f32_to_e4m3(f.y) << 8) |
           (f32_to_e4m3(f.z) << 16) | (f32_to_e4m3(f.w) << 24);
  }
  *(uint4*)(out + (size_t)t * 16) = make_uint4(w[0], w[1], w[2], w[3]);
}

__device__ __forceinline__ i32x8 tdm_group1(unsigned tensor_d0,
                                            unsigned tensor_d1,
                                            unsigned long long row_stride) {
  i32x8 g1;
  g1[0] = 0;
  g1[1] = (int)((tensor_d0 & 0xFFFFu) << 16);
  g1[2] = (int)(((tensor_d0 >> 16) & 0xFFFFu) |
                ((tensor_d1 & 0xFFFFu) << 16));
  g1[3] = (int)(((tensor_d1 >> 16) & 0xFFFFu) |
                (128u << 16));
  g1[4] = (int)(128u);
  g1[5] = (int)(unsigned)(row_stride & 0xFFFFFFFFu);
  g1[6] = (int)(unsigned)((row_stride >> 32) & 0xFFFFu);
  g1[7] = 0;
  return g1;
}

__device__ __forceinline__ void tdm_issue(unsigned lds_off,
                                          const unsigned char* gaddr,
                                          i32x8 g1) {
  unsigned long long ga = (unsigned long long)gaddr;
  u32x4 g0;
  g0[0] = 1u;
  g0[1] = lds_off;
  g0[2] = (unsigned)(ga & 0xFFFFFFFFu);
  g0[3] = (unsigned)((ga >> 32) & 0x01FFFFFFu)
          | (2u << 30);
  i32x4 z4 = {0, 0, 0, 0};
#if __has_include(<hip/amd_detail/amd_gfx1250_TDM.h>)
  i32x8 z8 = {0, 0, 0, 0, 0, 0, 0, 0};
  __builtin_amdgcn_tensor_load_to_lds(g0, g1, z4, z4, z8, 0);
#else
  __builtin_amdgcn_tensor_load_to_lds(g0, g1, z4, z4, 0);
#endif
}

__global__ __launch_bounds__(256) void fp8_gemm_wmma(
    const unsigned char* __restrict__ Aq,
    const unsigned char* __restrict__ Bq,
    float* __restrict__ C) {
  __shared__ __align__(16) unsigned char ldsA[2][128 * KSTEP];
  __shared__ __align__(16) unsigned char ldsB[2][128 * KSTEP];

  const int bn = blockIdx.x;
  const int bm = blockIdx.y;
  const int rowBase = bm * 128;
  const int colBase = bn * 128;

  const int tid   = threadIdx.x;
  const int wid   = tid >> 5;
  const int lane  = tid & 31;
  const int l15   = lane & 15;
  const int hi    = lane >> 4;
  const int waveM = wid >> 1;
  const int waveN = wid & 1;

  const v8f vzero = {0.f, 0.f, 0.f, 0.f, 0.f, 0.f, 0.f, 0.f};
  v8f acc[2][4];
#pragma unroll
  for (int mt = 0; mt < 2; ++mt)
#pragma unroll
    for (int nt = 0; nt < 4; ++nt) acc[mt][nt] = vzero;

  const unsigned ldsA0 = (unsigned)(unsigned long long)(const void*)ldsA[0];
  const unsigned ldsA1 = (unsigned)(unsigned long long)(const void*)ldsA[1];
  const unsigned ldsB0 = (unsigned)(unsigned long long)(const void*)ldsB[0];
  const unsigned ldsB1 = (unsigned)(unsigned long long)(const void*)ldsB[1];

  const unsigned char* aNext = Aq + (size_t)rowBase * GEMM_K;
  const unsigned char* bNext = Bq + (size_t)colBase * GEMM_K;
  const i32x8 g1A = tdm_group1(GEMM_K, GEMM_M, GEMM_K);
  const i32x8 g1B = tdm_group1(GEMM_K, GEMM_N, GEMM_K);

  const unsigned aFragOff = (unsigned)((waveM * 32 + l15) * KSTEP + hi * 16);
  const unsigned bFragOff = (unsigned)((waveN * 64 + l15) * KSTEP + hi * 16);

  auto compute = [&](const unsigned char* bufA, const unsigned char* bufB) {
    v16i afrag[2];
#pragma unroll
    for (int mt = 0; mt < 2; ++mt) {
      const unsigned char* base = bufA + aFragOff + (unsigned)(mt * 16 * KSTEP);
#pragma unroll
      for (int c = 0; c < 8; ++c) {
        u32x2 q = *(const u32x2*)(base + (c >> 1) * 32 + (c & 1) * 8);
        afrag[mt][2 * c]     = (int)q.x;
        afrag[mt][2 * c + 1] = (int)q.y;
      }
    }
    v16i bfrag[4];
#pragma unroll
    for (int nt = 0; nt < 4; ++nt) {
      const unsigned char* base = bufB + bFragOff + (unsigned)(nt * 16 * KSTEP);
#pragma unroll
      for (int j = 0; j < 4; ++j) {
        u32x4 q = *(const u32x4*)(base + j * 32);
        bfrag[nt][4 * j]     = (int)q.x;
        bfrag[nt][4 * j + 1] = (int)q.y;
        bfrag[nt][4 * j + 2] = (int)q.z;
        bfrag[nt][4 * j + 3] = (int)q.w;
      }
    }
#pragma unroll
    for (int mt = 0; mt < 2; ++mt)
#pragma unroll
      for (int nt = 0; nt < 4; ++nt)
        acc[mt][nt] = __builtin_amdgcn_wmma_f32_16x16x128_fp8_fp8(
            afrag[mt], bfrag[nt], (short)0, acc[mt][nt], false, false);
  };

  if (wid == 0) {
    tdm_issue(ldsA0, aNext, g1A);
    tdm_issue(ldsB0, bNext, g1B);
  }
  aNext += KSTEP;
  bNext += KSTEP;

#pragma unroll 1
  for (int i = 0; i < KSTEPS; i += 2) {
    if (wid == 0) {
      tdm_issue(ldsA1, aNext, g1A);
      tdm_issue(ldsB1, bNext, g1B);
      __builtin_amdgcn_s_wait_tensorcnt(2);
    }
    __syncthreads();
    compute(ldsA[0], ldsB[0]);
    __syncthreads();

    if (wid == 0) {
      if (i + 2 < KSTEPS) {
        tdm_issue(ldsA0, aNext + KSTEP, g1A);
        tdm_issue(ldsB0, bNext + KSTEP, g1B);
        __builtin_amdgcn_s_wait_tensorcnt(2);
      } else {
        __builtin_amdgcn_s_wait_tensorcnt(0);
      }
    }
    aNext += 2 * KSTEP;
    bNext += 2 * KSTEP;
    __syncthreads();
    compute(ldsA[1], ldsB[1]);
    __syncthreads();
  }

#pragma unroll
  for (int mt = 0; mt < 2; ++mt)
#pragma unroll
    for (int nt = 0; nt < 4; ++nt) {
      int gRow = rowBase + waveM * 32 + mt * 16 + hi * 8;
      int gCol = colBase + waveN * 64 + nt * 16 + l15;
      float* cp = C + (size_t)gRow * GEMM_N + gCol;
#pragma unroll
      for (int j = 0; j < 8; ++j)
        __builtin_nontemporal_store(acc[mt][nt][j], cp + (size_t)j * GEMM_N);
    }
}

extern "C" void kernel_launch(void* const* d_in, const int* in_sizes, int n_in,
                              void* d_out, int out_size, void* d_ws, size_t ws_size,
                              hipStream_t stream) {
  (void)in_sizes; (void)n_in; (void)out_size; (void)ws_size;
  const float* x = (const float*)d_in[0];
  const float* w = (const float*)d_in[1];
  unsigned char* xq = (unsigned char*)d_ws;
  unsigned char* wq = xq + (size_t)GEMM_M * GEMM_K;
  float* out = (float*)d_out;

  int n16x = (GEMM_M * GEMM_K) / 16;
  quantize_e4m3_kernel<<<(n16x + 255) / 256, 256, 0, stream>>>(x, xq, n16x);
  int n16w = (GEMM_N * GEMM_K) / 16;
  quantize_e4m3_kernel<<<(n16w + 255) / 256, 256, 0, stream>>>(w, wq, n16w);

  dim3 grid(GEMM_N / 128, GEMM_M / 128);
  fp8_gemm_wmma<<<grid, 256, 0, stream>>>(xq, wq, out);
}
